// VQuantizer_46703474376947
// MI455X (gfx1250) — matrix operand order repaired and checked by reading, not compiled, not run on hardware
//
#include <hip/hip_runtime.h>
#include <hip/hip_bf16.h>
#include <math.h>

// ---------------------------------------------------------------------------
// Model dims (match reference)
// ---------------------------------------------------------------------------
#define LAY   12
#define CH    512
#define NHEAD 8
#define HD    64          // CH / NHEAD
#define DFF   512
#define KCB   2048        // codebook size
#define NB    8           // batch
#define SEQ   256         // H*W
#define ROWS  2048        // SEQ * NB
#define LN_EPS 1e-5f

// ---------------------------------------------------------------------------
// WMMA types
// ---------------------------------------------------------------------------
typedef __attribute__((ext_vector_type(16))) __bf16 v16bf;
typedef __attribute__((ext_vector_type(8)))  float  v8f;

union FragBF { v16bf v; __bf16 e[16]; };
union FragF  { v8f  v; float  f[8];  };

__device__ __forceinline__ float gelu_exact(float x) {
    return 0.5f * x * (1.0f + erff(x * 0.7071067811865475f));
}

// ---------------------------------------------------------------------------
// Generic batched/strided WMMA GEMM:  C = act(scale * (A @ B[^T]) + bias)
//   A: M x K (f32, row stride lda); B: K x N normal -> B[k*ldb+n],
//   TRANSB -> B[n*ldb+k] (torch weights are (out,in)).
//   One wave computes a 32x32 output tile: 2 A-frags x 2 B-frags -> 4
//   independent v_wmma_f32_16x16x32_bf16 per K-step (f32 accumulation).
//   Fragment layouts used here (contiguous K, same order for A and B):
//     A 16x32 bf16: lane<16 row M=lane, K 0..15; lane>=16 K 16..31
//     B 32x16 bf16: lane<16 col N=lane, K 0..15; lane>=16 K 16..31
//     C/D f32:      VGPR r -> row r (+8 for lanes>=16), col = lane%16
// ---------------------------------------------------------------------------
struct GemmArgs {
    const float* A; const float* B; float* C; const float* bias;
    int M, N, K;
    int lda, ldb, ldc;
    long long aBase, bBase, cBase;
    long long aS1, aS2, bS1, bS2, cS1, cS2;
    int B2, act;
    float scale;
};

template <int TRANSB>
__global__ __launch_bounds__(128)
void wmma_gemm_kernel(GemmArgs g) {
    const int lane  = threadIdx.x;             // 0..31
    const int wave  = threadIdx.y;             // 0..3
    const int nTiles = (g.N + 31) >> 5;        // 32-wide tiles
    const int nTile  = blockIdx.x * 4 + wave;
    if (nTile >= nTiles) return;               // wave-uniform -> EXEC stays all-1s
    const int mTile = blockIdx.y;              // 32-high tiles
    const int z  = blockIdx.z;
    const int b1 = z / g.B2, b2 = z - b1 * g.B2;

    const float* Ab = g.A + g.aBase + (long long)b1 * g.aS1 + (long long)b2 * g.aS2;
    const float* Bb = g.B + g.bBase + (long long)b1 * g.bS1 + (long long)b2 * g.bS2;
    float*       Cb = g.C + g.cBase + (long long)b1 * g.cS1 + (long long)b2 * g.cS2;

    const int l16 = lane & 15;
    const int hi  = lane >> 4;                 // 0 or 1
    const int row0 = mTile * 32 + l16;         // A rows held by this lane
    const int row1 = row0 + 16;
    const int col0 = nTile * 32 + l16;         // B cols held by this lane
    const int col1 = col0 + 16;
    const int khA = hi ? 16 : 0;               // A-frag K pattern
    const int khB = hi ? 16 : 0;               // B-frag K pattern

    FragF acc00, acc01, acc10, acc11;
#pragma unroll
    for (int r = 0; r < 8; ++r) {
        acc00.f[r] = 0.0f; acc01.f[r] = 0.0f;
        acc10.f[r] = 0.0f; acc11.f[r] = 0.0f;
    }

    for (int k0 = 0; k0 < g.K; k0 += 32) {
        FragBF a0, a1, b0, b1;
        {
            const float* ap0 = Ab + (size_t)row0 * g.lda + k0 + khA;
            const float* ap1 = Ab + (size_t)row1 * g.lda + k0 + khA;
#pragma unroll
            for (int e = 0; e < 8; ++e) {
                a0.e[e]     = (__bf16)ap0[e];
                a0.e[8 + e] = (__bf16)ap0[8 + e];
                a1.e[e]     = (__bf16)ap1[e];
                a1.e[8 + e] = (__bf16)ap1[8 + e];
            }
        }
        if (TRANSB) {
            const float* bp0 = Bb + (size_t)col0 * g.ldb + k0 + khB;
            const float* bp1 = Bb + (size_t)col1 * g.ldb + k0 + khB;
#pragma unroll
            for (int e = 0; e < 16; ++e) {
                b0.e[e] = (__bf16)bp0[e];
                b1.e[e] = (__bf16)bp1[e];
            }
        } else {
            const float* bp = Bb + (size_t)(k0 + khB) * g.ldb;
#pragma unroll
            for (int e = 0; e < 16; ++e) {
                b0.e[e] = (__bf16)bp[(size_t)e * g.ldb + col0];
                b1.e[e] = (__bf16)bp[(size_t)e * g.ldb + col1];
            }
        }
        acc00.v = __builtin_amdgcn_wmma_f32_16x16x32_bf16(
                      false, a0.v, false, b0.v, (short)0, acc00.v, false, false);
        acc01.v = __builtin_amdgcn_wmma_f32_16x16x32_bf16(
                      false, a0.v, false, b1.v, (short)0, acc01.v, false, false);
        acc10.v = __builtin_amdgcn_wmma_f32_16x16x32_bf16(
                      false, a1.v, false, b0.v, (short)0, acc10.v, false, false);
        acc11.v = __builtin_amdgcn_wmma_f32_16x16x32_bf16(
                      false, a1.v, false, b1.v, (short)0, acc11.v, false, false);
    }

    const float bv0 = g.bias ? g.bias[col0] : 0.0f;
    const float bv1 = g.bias ? g.bias[col1] : 0.0f;
#pragma unroll
    for (int r = 0; r < 8; ++r) {
        int m0 = mTile * 32 + hi * 8 + r;      // rows of upper 16x32 strip
        int m1 = m0 + 16;                      // rows of lower strip
        float v00 = acc00.f[r] * g.scale + bv0;
        float v01 = acc01.f[r] * g.scale + bv1;
        float v10 = acc10.f[r] * g.scale + bv0;
        float v11 = acc11.f[r] * g.scale + bv1;
        if (g.act) {
            v00 = gelu_exact(v00); v01 = gelu_exact(v01);
            v10 = gelu_exact(v10); v11 = gelu_exact(v11);
        }
        Cb[(size_t)m0 * g.ldc + col0] = v00;
        Cb[(size_t)m0 * g.ldc + col1] = v01;
        Cb[(size_t)m1 * g.ldc + col0] = v10;
        Cb[(size_t)m1 * g.ldc + col1] = v11;
    }
}

// ---------------------------------------------------------------------------
// 2D sinusoidal positional encoding, computed analytically
// ---------------------------------------------------------------------------
__device__ __forceinline__ float pe_val(int s, int c) {
    int hh = s >> 4, ww = s & 15;
    int pos, cc;
    if (c < 256) { pos = hh; cc = c; } else { pos = ww; cc = c - 256; }
    int i = cc >> 1;
    float div = expf(-(float)(2 * i) * (9.210340371976184f / 256.0f));
    float ang = (float)pos * div;
    return (cc & 1) ? cosf(ang) : sinf(ang);
}

// x[(s*8+b)*512+c] = latents[b,c,hh,ww] + pe(s,c)
__global__ __launch_bounds__(256)
void add_pe_latents_kernel(const float* __restrict__ lat, float* __restrict__ x) {
    int idx = blockIdx.x * 256 + threadIdx.x;          // < ROWS*CH
    int row = idx >> 9, c = idx & 511;
    int s = row >> 3, b = row & 7;
    x[idx] = lat[(size_t)b * (CH * SEQ) + (size_t)c * SEQ + s] + pe_val(s, c);
}

// pq = zs + pe ; write into both MEM (cross-attn memory) and X (decoder state)
__global__ __launch_bounds__(256)
void add_pe_zs_kernel(float* __restrict__ x, float* __restrict__ mem) {
    int idx = blockIdx.x * 256 + threadIdx.x;
    int row = idx >> 9, c = idx & 511;
    int s = row >> 3;
    float v = x[idx] + pe_val(s, c);
    x[idx] = v; mem[idx] = v;
}

// ---------------------------------------------------------------------------
// Fused residual-add + LayerNorm over rows of 512:  x = LN(x + delta) * w + b
// ---------------------------------------------------------------------------
__global__ __launch_bounds__(256)
void add_ln_kernel(float* __restrict__ x, const float* __restrict__ delta,
                   const float* __restrict__ w, const float* __restrict__ b) {
    __shared__ float sh[256], sh2[256];
    size_t row = blockIdx.x;
    int t = threadIdx.x;
    float* xp = x + row * CH;
    const float* dp = delta + row * CH;
    float v0 = xp[t] + dp[t];
    float v1 = xp[t + 256] + dp[t + 256];
    sh[t] = v0 + v1; sh2[t] = v0 * v0 + v1 * v1;
    __syncthreads();
    for (int s = 128; s > 0; s >>= 1) {
        if (t < s) { sh[t] += sh[t + s]; sh2[t] += sh2[t + s]; }
        __syncthreads();
    }
    float mean = sh[0] * (1.0f / CH);
    float var  = sh2[0] * (1.0f / CH) - mean * mean;
    float r = rsqrtf(var + LN_EPS);
    xp[t]       = (v0 - mean) * r * w[t]       + b[t];
    xp[t + 256] = (v1 - mean) * r * w[t + 256] + b[t + 256];
}

// ---------------------------------------------------------------------------
// Row softmax (block per row). divByMax: softmax(x / rowmax(x)) (VQ soft path)
// ---------------------------------------------------------------------------
__global__ __launch_bounds__(256)
void softmax_rows_kernel(float* __restrict__ data, int cols, int divByMax) {
    __shared__ float sh[256];
    size_t row = blockIdx.x;
    float* p = data + row * (size_t)cols;
    int t = threadIdx.x;
    float mx = -3.4e38f;
    for (int c = t; c < cols; c += 256) mx = fmaxf(mx, p[c]);
    sh[t] = mx; __syncthreads();
    for (int s = 128; s > 0; s >>= 1) { if (t < s) sh[t] = fmaxf(sh[t], sh[t + s]); __syncthreads(); }
    mx = sh[0]; __syncthreads();
    float scl = divByMax ? (1.0f / mx) : 1.0f;
    float shift = mx * scl;
    float sum = 0.0f;
    for (int c = t; c < cols; c += 256) { float e = expf(p[c] * scl - shift); p[c] = e; sum += e; }
    sh[t] = sum; __syncthreads();
    for (int s = 128; s > 0; s >>= 1) { if (t < s) sh[t] += sh[t + s]; __syncthreads(); }
    float inv = 1.0f / sh[0];
    for (int c = t; c < cols; c += 256) p[c] *= inv;
}

// ---------------------------------------------------------------------------
// VQ helper kernels
// ---------------------------------------------------------------------------
__global__ __launch_bounds__(256)
void rowsumsq_kernel(const float* __restrict__ x, float* __restrict__ out) {
    __shared__ float sh[256];
    size_t row = blockIdx.x;
    const float* p = x + row * CH;
    int t = threadIdx.x;
    float a = p[t], bb = p[t + 256];
    sh[t] = a * a + bb * bb; __syncthreads();
    for (int s = 128; s > 0; s >>= 1) { if (t < s) sh[t] += sh[t + s]; __syncthreads(); }
    if (t == 0) out[row] = sh[0];
}

// csq[k] = sum_c codebook[c,k]^2  (codebook is (C, K) row-major)
__global__ __launch_bounds__(256)
void colsumsq_kernel(const float* __restrict__ cb, float* __restrict__ out) {
    int k = blockIdx.x * 256 + threadIdx.x;            // < KCB
    float s = 0.0f;
    for (int c = 0; c < CH; ++c) { float v = cb[(size_t)c * KCB + k]; s += v * v; }
    out[k] = s;
}

// d = xsq[m] + csq[n] + (-2 * x@cb already in d)
__global__ __launch_bounds__(256)
void dist_finish_kernel(float* __restrict__ d, const float* __restrict__ xsq,
                        const float* __restrict__ csq) {
    size_t idx = (size_t)blockIdx.x * 256 + threadIdx.x;   // < ROWS*KCB
    int m = (int)(idx >> 11), n = (int)(idx & 2047);
    d[idx] += xsq[m] + csq[n];
}

__global__ __launch_bounds__(256)
void argmin_rows_kernel(const float* __restrict__ d, int cols, int* __restrict__ outIdx) {
    __shared__ float shv[256]; __shared__ int shi[256];
    size_t row = blockIdx.x;
    const float* p = d + row * (size_t)cols;
    int t = threadIdx.x;
    float best = 3.4e38f; int bi = 0;
    for (int c = t; c < cols; c += 256) { float v = p[c]; if (v < best) { best = v; bi = c; } }
    shv[t] = best; shi[t] = bi; __syncthreads();
    for (int s = 128; s > 0; s >>= 1) {
        if (t < s) {
            if (shv[t + s] < shv[t] || (shv[t + s] == shv[t] && shi[t + s] < shi[t])) {
                shv[t] = shv[t + s]; shi[t] = shi[t + s];
            }
        }
        __syncthreads();
    }
    if (t == 0) outIdx[row] = shi[0];
}

// zq[row,c] = cb[c, nearest[row]]
__global__ __launch_bounds__(256)
void zq_gather_kernel(const float* __restrict__ cb, const int* __restrict__ nearest,
                      float* __restrict__ out) {
    size_t idx = (size_t)blockIdx.x * 256 + threadIdx.x;   // < ROWS*CH
    int row = (int)(idx >> 9), c = (int)(idx & 511);
    out[idx] = cb[(size_t)c * KCB + nearest[row]];
}

// codes[b, s] = nearest[s*8+b]  (stored as float in output)
__global__ __launch_bounds__(256)
void codes_kernel(const int* __restrict__ nearest, float* __restrict__ out) {
    int idx = blockIdx.x * 256 + threadIdx.x;              // < ROWS
    int b = idx >> 8, s = idx & 255;
    out[idx] = (float)nearest[s * NB + b];
}

// cw_t[k, c] = cb[c, k]
__global__ __launch_bounds__(256)
void cwt_kernel(const float* __restrict__ cb, float* __restrict__ out) {
    size_t idx = (size_t)blockIdx.x * 256 + threadIdx.x;   // < KCB*CH
    int k = (int)(idx >> 9), c = (int)(idx & 511);
    out[idx] = cb[(size_t)c * KCB + k];
}

// de[b, c, s] = y[(s*8+b)*512 + c]
__global__ __launch_bounds__(256)
void de_output_kernel(const float* __restrict__ y, float* __restrict__ out) {
    size_t idx = (size_t)blockIdx.x * 256 + threadIdx.x;   // < NB*CH*SEQ
    int b = (int)(idx / (CH * SEQ));
    int rem = (int)(idx - (size_t)b * (CH * SEQ));
    int c = rem >> 8, s = rem & 255;
    out[idx] = y[((size_t)s * NB + b) * CH + c];
}

// ---------------------------------------------------------------------------
// Host-side orchestration
// ---------------------------------------------------------------------------
static void gemm(hipStream_t st,
                 const float* A, const float* B, float* C, const float* bias,
                 int M, int N, int K, int lda, int ldb, int ldc,
                 long long aBase, long long bBase, long long cBase,
                 long long aS1, long long aS2, long long bS1, long long bS2,
                 long long cS1, long long cS2,
                 int B1, int B2, int transB, int act, float scale) {
    GemmArgs g = {A, B, C, bias, M, N, K, lda, ldb, ldc,
                  aBase, bBase, cBase, aS1, aS2, bS1, bS2, cS1, cS2,
                  B2, act, scale};
    int nT = (N + 31) / 32, mT = (M + 31) / 32;
    dim3 grd((nT + 3) / 4, mT, B1 * B2);
    dim3 blk(32, 4, 1);
    if (transB) wmma_gemm_kernel<1><<<grd, blk, 0, st>>>(g);
    else        wmma_gemm_kernel<0><<<grd, blk, 0, st>>>(g);
}

// Batched attention using packed QKV buffer (ROWS x 1536: [q|k|v], row = s*8+b)
static void attention(hipStream_t st, float* QKV, float* SCR, float* CTX) {
    // scores[b,h,s,t] = (1/8) * sum_d Q[s,d] K[t,d]
    gemm(st, QKV, QKV, SCR, nullptr,
         SEQ, SEQ, HD, NB * 3 * CH, NB * 3 * CH, SEQ,
         0, CH, 0,
         3 * CH, HD, 3 * CH, HD, (long long)NHEAD * SEQ * SEQ, (long long)SEQ * SEQ,
         NB, NHEAD, /*transB=*/1, 0, 0.125f);
    softmax_rows_kernel<<<NB * NHEAD * SEQ, 256, 0, st>>>(SCR, SEQ, 0);
    // ctx[s, b*512 + h*64 + d] = sum_t P[s,t] V[t,d]
    gemm(st, SCR, QKV, CTX, nullptr,
         SEQ, HD, SEQ, SEQ, NB * 3 * CH, NB * CH,
         0, 2 * CH, 0,
         (long long)NHEAD * SEQ * SEQ, (long long)SEQ * SEQ, 3 * CH, HD, CH, HD,
         NB, NHEAD, /*transB=*/0, 0, 1.0f);
}

extern "C" void kernel_launch(void* const* d_in, const int* in_sizes, int n_in,
                              void* d_out, int out_size, void* d_ws, size_t ws_size,
                              hipStream_t stream) {
    (void)in_sizes; (void)n_in; (void)out_size; (void)ws_size;

    const float* lat        = (const float*)d_in[0];
    const float* e_qkv_w    = (const float*)d_in[1];
    const float* e_qkv_b    = (const float*)d_in[2];
    const float* e_out_w    = (const float*)d_in[3];
    const float* e_out_b    = (const float*)d_in[4];
    const float* e_f1_w     = (const float*)d_in[5];
    const float* e_f1_b     = (const float*)d_in[6];
    const float* e_f2_w     = (const float*)d_in[7];
    const float* e_f2_b     = (const float*)d_in[8];
    const float* e_l1_w     = (const float*)d_in[9];
    const float* e_l1_b     = (const float*)d_in[10];
    const float* e_l2_w     = (const float*)d_in[11];
    const float* e_l2_b     = (const float*)d_in[12];
    const float* d_sa_qkv_w = (const float*)d_in[13];
    const float* d_sa_qkv_b = (const float*)d_in[14];
    const float* d_sa_out_w = (const float*)d_in[15];
    const float* d_sa_out_b = (const float*)d_in[16];
    const float* d_ca_qkv_w = (const float*)d_in[17];
    const float* d_ca_qkv_b = (const float*)d_in[18];
    const float* d_ca_out_w = (const float*)d_in[19];
    const float* d_ca_out_b = (const float*)d_in[20];
    const float* d_f1_w     = (const float*)d_in[21];
    const float* d_f1_b     = (const float*)d_in[22];
    const float* d_f2_w     = (const float*)d_in[23];
    const float* d_f2_b     = (const float*)d_in[24];
    const float* d_l1_w     = (const float*)d_in[25];
    const float* d_l1_b     = (const float*)d_in[26];
    const float* d_l2_w     = (const float*)d_in[27];
    const float* d_l2_b     = (const float*)d_in[28];
    const float* d_l3_w     = (const float*)d_in[29];
    const float* d_l3_b     = (const float*)d_in[30];
    const float* cb         = (const float*)d_in[31];   // (C, K)

    float* out = (float*)d_out;
    const long long DE_OFF    = 0;
    const long long CODES_OFF = (long long)NB * CH * SEQ;                 // 1048576
    const long long ZS_OFF    = CODES_OFF + (long long)NB * SEQ;          // 1050624
    const long long ZQ_OFF    = ZS_OFF + (long long)ROWS * CH;            // 2099200
    const long long SOFT_OFF  = ZQ_OFF + (long long)ROWS * CH;            // 3147776
    const long long CWT_OFF   = SOFT_OFF + (long long)ROWS * CH;          // 4196352

    float* ws = (float*)d_ws;
    float* X    = ws;                                   // ROWS*CH
    float* MEM  = X   + (size_t)ROWS * CH;              // ROWS*CH
    float* CTX  = MEM + (size_t)ROWS * CH;              // ROWS*CH (attn ctx / ffn hidden)
    float* QKV  = CTX + (size_t)ROWS * CH;              // ROWS*3*CH (also sublayer tmp)
    float* SCR  = QKV + (size_t)ROWS * 3 * CH;          // 64*SEQ*SEQ (scores / VQ dist)
    float* XSQ  = SCR + (size_t)NB * NHEAD * SEQ * SEQ; // ROWS
    float* CSQ  = XSQ + ROWS;                           // KCB
    int*   NEAR = (int*)(CSQ + KCB);                    // ROWS ints

    const int EB = (ROWS * CH) / 256;                   // 4096 blocks for elementwise

    // ---- encoder input: x = latents(transposed) + PE ----
    add_pe_latents_kernel<<<EB, 256, 0, stream>>>(lat, X);

    // ---- encoder layers ----
    for (int l = 0; l < LAY; ++l) {
        long long wq = (long long)l * 3 * CH * CH, bq = (long long)l * 3 * CH;
        long long wo = (long long)l * CH * CH,     bo = (long long)l * CH;
        // QKV projection: (2048,512) @ (512,1536) + b
        gemm(stream, X, e_qkv_w + wq, QKV, e_qkv_b + bq,
             ROWS, 3 * CH, CH, CH, CH, 3 * CH,
             0, 0, 0, 0, 0, 0, 0, 0, 0, 1, 1, 1, 0, 1.0f);
        attention(stream, QKV, SCR, CTX);
        // out projection -> tmp (reuse QKV buffer)
        gemm(stream, CTX, e_out_w + wo, QKV, e_out_b + bo,
             ROWS, CH, CH, CH, CH, CH,
             0, 0, 0, 0, 0, 0, 0, 0, 0, 1, 1, 1, 0, 1.0f);
        add_ln_kernel<<<ROWS, 256, 0, stream>>>(X, QKV, e_l1_w + bo, e_l1_b + bo);
        // FFN: gelu(x@W1^T+b1)@W2^T+b2
        gemm(stream, X, e_f1_w + wo, CTX, e_f1_b + bo,
             ROWS, DFF, CH, CH, CH, DFF,
             0, 0, 0, 0, 0, 0, 0, 0, 0, 1, 1, 1, /*gelu*/1, 1.0f);
        gemm(stream, CTX, e_f2_w + wo, QKV, e_f2_b + bo,
             ROWS, CH, DFF, DFF, DFF, CH,
             0, 0, 0, 0, 0, 0, 0, 0, 0, 1, 1, 1, 0, 1.0f);
        add_ln_kernel<<<ROWS, 256, 0, stream>>>(X, QKV, e_l2_w + bo, e_l2_b + bo);
    }

    // ---- zs output ----
    hipMemcpyAsync(out + ZS_OFF, X, sizeof(float) * ROWS * CH,
                   hipMemcpyDeviceToDevice, stream);

    // ---- VQ: d = |x|^2 + |c|^2 - 2 x@cb ----
    rowsumsq_kernel<<<ROWS, 256, 0, stream>>>(X, XSQ);
    colsumsq_kernel<<<KCB / 256, 256, 0, stream>>>(cb, CSQ);
    gemm(stream, X, cb, SCR, nullptr,
         ROWS, KCB, CH, CH, KCB, KCB,
         0, 0, 0, 0, 0, 0, 0, 0, 0, 1, 1, /*transB=*/0, 0, -2.0f);
    dist_finish_kernel<<<(ROWS * KCB) / 256, 256, 0, stream>>>(SCR, XSQ, CSQ);
    argmin_rows_kernel<<<ROWS, 256, 0, stream>>>(SCR, KCB, NEAR);
    zq_gather_kernel<<<EB, 256, 0, stream>>>(cb, NEAR, out + ZQ_OFF);
    codes_kernel<<<ROWS / 256, 256, 0, stream>>>(NEAR, out + CODES_OFF);
    // soft = softmax(d / rowmax(d)) @ cw_t
    softmax_rows_kernel<<<ROWS, 256, 0, stream>>>(SCR, KCB, 1);
    gemm(stream, SCR, cb, out + SOFT_OFF, nullptr,
         ROWS, CH, KCB, KCB, KCB, CH,
         0, 0, 0, 0, 0, 0, 0, 0, 0, 1, 1, /*transB=*/1, 0, 1.0f);
    cwt_kernel<<<EB, 256, 0, stream>>>(cb, out + CWT_OFF);

    // ---- decoder input: pq = zs + PE; mem = pq ----
    add_pe_zs_kernel<<<EB, 256, 0, stream>>>(X, MEM);

    // ---- decoder layers ----
    for (int l = 0; l < LAY; ++l) {
        long long wq = (long long)l * 3 * CH * CH, bq = (long long)l * 3 * CH;
        long long wo = (long long)l * CH * CH,     bo = (long long)l * CH;
        // self-attention
        gemm(stream, X, d_sa_qkv_w + wq, QKV, d_sa_qkv_b + bq,
             ROWS, 3 * CH, CH, CH, CH, 3 * CH,
             0, 0, 0, 0, 0, 0, 0, 0, 0, 1, 1, 1, 0, 1.0f);
        attention(stream, QKV, SCR, CTX);
        gemm(stream, CTX, d_sa_out_w + wo, QKV, d_sa_out_b + bo,
             ROWS, CH, CH, CH, CH, CH,
             0, 0, 0, 0, 0, 0, 0, 0, 0, 1, 1, 1, 0, 1.0f);
        add_ln_kernel<<<ROWS, 256, 0, stream>>>(X, QKV, d_l1_w + bo, d_l1_b + bo);
        // cross-attention: q from X, k/v from MEM, packed into same QKV buffer
        gemm(stream, X, d_ca_qkv_w + wq, QKV, d_ca_qkv_b + bq,
             ROWS, CH, CH, CH, CH, 3 * CH,
             0, 0, 0, 0, 0, 0, 0, 0, 0, 1, 1, 1, 0, 1.0f);
        gemm(stream, MEM, d_ca_qkv_w + wq + (long long)CH * CH, QKV,
             d_ca_qkv_b + bq + CH,
             ROWS, 2 * CH, CH, CH, CH, 3 * CH,
             0, 0, /*cBase=*/CH, 0, 0, 0, 0, 0, 0, 1, 1, 1, 0, 1.0f);
        attention(stream, QKV, SCR, CTX);
        gemm(stream, CTX, d_ca_out_w + wo, QKV, d_ca_out_b + bo,
             ROWS, CH, CH, CH, CH, CH,
             0, 0, 0, 0, 0, 0, 0, 0, 0, 1, 1, 1, 0, 1.0f);
        add_ln_kernel<<<ROWS, 256, 0, stream>>>(X, QKV, d_l2_w + bo, d_l2_b + bo);
        // FFN
        gemm(stream, X, d_f1_w + wo, CTX, d_f1_b + bo,
             ROWS, DFF, CH, CH, CH, DFF,
             0, 0, 0, 0, 0, 0, 0, 0, 0, 1, 1, 1, /*gelu*/1, 1.0f);
        gemm(stream, CTX, d_f2_w + wo, QKV, d_f2_b + bo,
             ROWS, CH, DFF, DFF, DFF, CH,
             0, 0, 0, 0, 0, 0, 0, 0, 0, 1, 1, 1, 0, 1.0f);
        add_ln_kernel<<<ROWS, 256, 0, stream>>>(X, QKV, d_l3_w + bo, d_l3_b + bo);
    }

    // ---- de output: (n, c, h, w) ----
    de_output_kernel<<<EB, 256, 0, stream>>>(X, out + DE_OFF);
}
